// ARCModel_57346403336342
// MI455X (gfx1250) — matrix operand order repaired and checked by reading, not compiled, not run on hardware
//
#include <hip/hip_runtime.h>
#include <hip/hip_bf16.h>
#include <math.h>

// ---------------------------------------------------------------------------
// Types / constants
// ---------------------------------------------------------------------------
typedef __bf16 bf16_t;
typedef __bf16 bf16x16 __attribute__((ext_vector_type(16)));
typedef __bf16 bf16x8  __attribute__((ext_vector_type(8)));
typedef float  v8f     __attribute__((ext_vector_type(8)));

#define T_TOK   18432           // 8*48*48 tokens
#define DIM_    768
#define HEADS_  12
#define DH_     64
#define DEPTH_  6
#define FFI_    2048
#define SEQN    48
#define SQRTD   27.712812921102035f          // sqrt(768)
#define ALPHAFS (SQRTD / 6.0f)               // Scale forward factor
#define EPS_    1e-12f

// ---------------------------------------------------------------------------
// WMMA helpers (CDNA5 gfx1250, wave32)
// ---------------------------------------------------------------------------
__device__ __forceinline__ v8f wmma_bf16(bf16x16 a, bf16x16 b, v8f c) {
  return __builtin_amdgcn_wmma_f32_16x16x32_bf16(false, a, false, b,
                                                 (short)0, c, false, false);
}

// A fragment (16x32, M x K): lane m (0-15) holds K {0..7,16..23}, lanes 16-31
// hold K {8..15,24..31} for the same rows.  base points at (row0, k0) in a
// row-major [rows][ld] bf16 tile; ld, offsets multiples of 8 (16B aligned).
__device__ __forceinline__ bf16x16 frag_a(const bf16_t* base, int ld,
                                          int half, int r) {
  const bf16_t* p = base + r * ld + half * 8;
  bf16x8 c0 = *(const bf16x8*)(p);
  bf16x8 c1 = *(const bf16x8*)(p + 16);
  bf16x16 f;
#pragma unroll
  for (int i = 0; i < 8; i++) { f[i] = c0[i]; f[i + 8] = c1[i]; }
  return f;
}

// B fragment (32x16, K x N) sourced from a row-major [N][K] tile (i.e. B^T):
// lane n (0-15) holds K {0..7,16..23}, lanes 16-31 hold K {8..15,24..31}.
__device__ __forceinline__ bf16x16 frag_b(const bf16_t* base, int ld,
                                          int half, int r) {
  const bf16_t* p = base + r * ld + half * 8;
  bf16x8 c0 = *(const bf16x8*)(p);
  bf16x8 c1 = *(const bf16x8*)(p + 16);
  bf16x16 f;
#pragma unroll
  for (int i = 0; i < 8; i++) { f[i] = c0[i]; f[i + 8] = c1[i]; }
  return f;
}

// CDNA5 async DMA: memory -> LDS, 16B per lane, tracked by ASYNCcnt.
// %0 = VGPR with LDS byte offset (addrspace(3) offset = low 32 bits of the
// flat shared-aperture address), %1 = 64-bit global address.
__device__ __forceinline__ void async_copy_b128(void* lds_ptr,
                                                const void* gptr) {
  unsigned lds_off = (unsigned)(unsigned long long)lds_ptr;
  asm volatile("global_load_async_to_lds_b128 %0, %1, off"
               :: "v"(lds_off), "v"(gptr)
               : "memory");
}
__device__ __forceinline__ void wait_async_le(int pending) {
  if (pending) asm volatile("s_wait_asynccnt 0x4" ::: "memory");
  else         asm volatile("s_wait_asynccnt 0x0" ::: "memory");
}

// 256-thread block reduction (sum)
__device__ __forceinline__ float block_reduce_256(float v, float* red) {
  int tid = threadIdx.x;
  red[tid] = v;
  __syncthreads();
  for (int s = 128; s > 0; s >>= 1) {
    if (tid < s) red[tid] += red[tid + s];
    __syncthreads();
  }
  float r = red[0];
  __syncthreads();
  return r;
}

// ---------------------------------------------------------------------------
// Weight row L2-normalization  (rows of length K; one block per row)
// ---------------------------------------------------------------------------
__global__ __launch_bounds__(256)
void rownorm_bf16_kernel(const float* __restrict__ src, bf16_t* __restrict__ dst,
                         int K) {
  __shared__ float red[256];
  size_t row = blockIdx.x;
  int tid = threadIdx.x;
  const float* s = src + row * (size_t)K;
  float ss = 0.f;
  for (int d = tid; d < K; d += 256) { float v = s[d]; ss += v * v; }
  ss = block_reduce_256(ss, red);
  float inv = 1.f / fmaxf(sqrtf(ss), EPS_);
  bf16_t* o = dst + row * (size_t)K;
  for (int d = tid; d < K; d += 256) o[d] = (bf16_t)(s[d] * inv);
}

__global__ __launch_bounds__(256)
void rownorm_f32_kernel(const float* __restrict__ src, float* __restrict__ dst,
                        int K) {
  __shared__ float red[256];
  size_t row = blockIdx.x;
  int tid = threadIdx.x;
  const float* s = src + row * (size_t)K;
  float ss = 0.f;
  for (int d = tid; d < K; d += 256) { float v = s[d]; ss += v * v; }
  ss = block_reduce_256(ss, red);
  float inv = 1.f / fmaxf(sqrtf(ss), EPS_);
  float* o = dst + row * (size_t)K;
  for (int d = tid; d < K; d += 256) o[d] = s[d] * inv;
}

// ---------------------------------------------------------------------------
// Embedding: x = l2n(ce + s_task*(te-ce)); one block per token
// ---------------------------------------------------------------------------
__global__ __launch_bounds__(256)
void embed_kernel(const int* __restrict__ inp, const int* __restrict__ task,
                  const float* __restrict__ wcol, const float* __restrict__ wtask,
                  const float* __restrict__ s_task,
                  float* __restrict__ x, bf16_t* __restrict__ xb) {
  __shared__ float red[256];
  size_t t = blockIdx.x;
  int tid = threadIdx.x;
  int c = inp[t];
  int ti = task[t / 2304];
  float v[3];
  float ss = 0.f;
#pragma unroll
  for (int k = 0; k < 3; k++) {
    int d = tid + k * 256;
    float ce = wcol[(size_t)c * DIM_ + d];
    float te = wtask[(size_t)ti * DIM_ + d];
    float u = ce + s_task[d] * (te - ce);
    v[k] = u;
    ss += u * u;
  }
  ss = block_reduce_256(ss, red);
  float inv = 1.f / fmaxf(sqrtf(ss), EPS_);
#pragma unroll
  for (int k = 0; k < 3; k++) {
    int d = tid + k * 256;
    float u = v[k] * inv;
    x[t * DIM_ + d] = u;
    xb[t * DIM_ + d] = (bf16_t)u;
  }
}

// ---------------------------------------------------------------------------
// Tiled bf16 WMMA GEMM:  C[M,N] = A[M,K] * W[N,K]^T
// block tile 128x128, K step 32; 8 waves, each wave 32x64 (2x4 WMMA frags).
// Double-buffered LDS, filled by GLOBAL_LOAD_ASYNC_TO_LDS_B128 (ASYNCcnt):
// each wave issues 4 async 512B copies per K-tile; while computing tile i the
// DMA engine fills tile i+1 (s_wait_asynccnt 4 retires only tile i, since
// async loads complete in order).
// EPI: 0 = store bf16
//      1 = store bf16 silu(acc * cscale[n] * sqrt(D))         (gate path)
//      2 = store bf16 acc * cscale[n] * gate[m,n]             (hidden path)
//      3 = store f32
//      4 = accumulate f32 (outf += acc)
// ---------------------------------------------------------------------------
template <int EPI>
__global__ __launch_bounds__(256)
void gemm_kernel(const bf16_t* __restrict__ A, const bf16_t* __restrict__ Bw,
                 int M, int N, int K,
                 bf16_t* outb, float* outf,
                 const float* cscale, const bf16_t* gate) {
  __shared__ bf16_t As[2][128 * 32];
  __shared__ bf16_t Bs[2][128 * 32];
  int bn = blockIdx.x, bm = blockIdx.y;
  int tid = threadIdx.x;
  int w = tid >> 5, lane = tid & 31;
  int wm = w & 3, wn = w >> 2;     // wave grid 4(M) x 2(N)
  int half = lane >> 4, r = lane & 15;

  v8f acc[2][4];
#pragma unroll
  for (int i = 0; i < 2; i++)
#pragma unroll
    for (int j = 0; j < 4; j++)
#pragma unroll
      for (int e = 0; e < 8; e++) acc[i][j][e] = 0.f;

  size_t row0A = (size_t)bm * 128;
  size_t row0B = (size_t)bn * 128;

  // Issue one K-tile's async copies (2 x 16B chunks per operand per thread).
  auto issue_tile = [&](int kt, int buf) {
#pragma unroll
    for (int cc = 0; cc < 2; cc++) {
      int chunk = tid + cc * 256;
      int rrow = chunk >> 2, c = chunk & 3;
      async_copy_b128(&As[buf][rrow * 32 + c * 8],
                      &A[(row0A + rrow) * (size_t)K + kt + c * 8]);
      async_copy_b128(&Bs[buf][rrow * 32 + c * 8],
                      &Bw[(row0B + rrow) * (size_t)K + kt + c * 8]);
    }
  };

  int nk = K >> 5;
  issue_tile(0, 0);
  for (int it = 0; it < nk; it++) {
    int cur = it & 1;
    int more = (it + 1 < nk);
    if (more) issue_tile((it + 1) << 5, cur ^ 1);
    wait_async_le(more);           // tile `it` resident in LDS (this wave)
    __syncthreads();               // ... and for all waves of the block

    bf16x16 fa[2], fb[4];
#pragma unroll
    for (int i = 0; i < 2; i++)
      fa[i] = frag_a(&As[cur][(wm * 32 + i * 16) * 32], 32, half, r);
#pragma unroll
    for (int j = 0; j < 4; j++)
      fb[j] = frag_b(&Bs[cur][(wn * 64 + j * 16) * 32], 32, half, r);
#pragma unroll
    for (int i = 0; i < 2; i++)
#pragma unroll
      for (int j = 0; j < 4; j++)
        acc[i][j] = wmma_bf16(fa[i], fb[j], acc[i][j]);
    __syncthreads();               // block done reading buf before reuse
  }

  // Epilogue
#pragma unroll
  for (int i = 0; i < 2; i++) {
#pragma unroll
    for (int j = 0; j < 4; j++) {
      int gn = bn * 128 + wn * 64 + j * 16 + r;
#pragma unroll
      for (int vi = 0; vi < 8; vi++) {
        int gm = bm * 128 + wm * 32 + i * 16 + vi + half * 8;
        size_t o = (size_t)gm * N + gn;
        float v = acc[i][j][vi];
        if constexpr (EPI == 0) {
          outb[o] = (bf16_t)v;
        } else if constexpr (EPI == 1) {
          float g = v * cscale[gn] * SQRTD;
          outb[o] = (bf16_t)(g / (1.f + __expf(-g)));    // silu
        } else if constexpr (EPI == 2) {
          outb[o] = (bf16_t)(v * cscale[gn] * (float)gate[o]);
        } else if constexpr (EPI == 3) {
          outf[o] = v;
        } else {
          outf[o] += v;
        }
      }
    }
  }
}

// ---------------------------------------------------------------------------
// Per-(token,head) l2norm + s_qk*sqrt(D) scaling of q and k, in place.
// One wave per (token,head); 2 elements per lane.
// ---------------------------------------------------------------------------
__global__ __launch_bounds__(32)
void qkprep_kernel(bf16_t* __restrict__ qb, bf16_t* __restrict__ kb,
                   const float* __restrict__ s_qk) {
  size_t g = blockIdx.x;
  int h = (int)(g % HEADS_);
  size_t t = g / HEADS_;
  int lane = threadIdx.x;
  size_t o = t * DIM_ + h * DH_ + lane * 2;
  float q0 = (float)qb[o], q1 = (float)qb[o + 1];
  float k0 = (float)kb[o], k1 = (float)kb[o + 1];
  float qs = q0 * q0 + q1 * q1;
  float ks = k0 * k0 + k1 * k1;
#pragma unroll
  for (int m = 16; m >= 1; m >>= 1) {
    qs += __shfl_xor(qs, m, 32);
    ks += __shfl_xor(ks, m, 32);
  }
  float qinv = 1.f / fmaxf(sqrtf(qs), EPS_);
  float kinv = 1.f / fmaxf(sqrtf(ks), EPS_);
  float s0 = s_qk[h * DH_ + lane * 2] * SQRTD;
  float s1 = s_qk[h * DH_ + lane * 2 + 1] * SQRTD;
  qb[o] = (bf16_t)(q0 * qinv * s0);
  qb[o + 1] = (bf16_t)(q1 * qinv * s1);
  kb[o] = (bf16_t)(k0 * kinv * s0);
  kb[o + 1] = (bf16_t)(k1 * kinv * s1);
}

// ---------------------------------------------------------------------------
// Axial attention, one wave per (sequence, head).
// dir 0: sequence along J (token stride 1); dir 1: along I (stride 48).
// RoPE applied on load (position = index within sequence).
// scores (48x48) via WMMA bf16, equality-mask, *sqrt(64), softmax,
// P -> bf16 (K padded to 64), AV via WMMA with V stored transposed.
// ---------------------------------------------------------------------------
__global__ __launch_bounds__(32)
void attn_kernel(const bf16_t* __restrict__ qb, const bf16_t* __restrict__ kb,
                 const bf16_t* __restrict__ vb, const int* __restrict__ inp,
                 bf16_t* __restrict__ ob, int dir) {
  __shared__ bf16_t qs[SEQN * 64];
  __shared__ bf16_t ks[SEQN * 64];
  __shared__ bf16_t vt[64 * 64];       // [dh][key], keys 48..63 zero pad
  __shared__ float  sc[SEQN * SEQN];
  __shared__ bf16_t ps[SEQN * 64];     // softmax probs, cols 48..63 zero pad
  __shared__ int    mk[SEQN];

  int sh = blockIdx.x;
  int h = sh % HEADS_;
  int s = sh / HEADS_;
  int lane = threadIdx.x;

  size_t base;
  size_t stride;
  if (dir == 0) { base = (size_t)s * SEQN; stride = 1; }
  else { base = (size_t)(s / SEQN) * 2304 + (size_t)(s % SEQN); stride = SEQN; }

  for (int i = lane; i < 64 * 64; i += 32) vt[i] = (bf16_t)0.f;
  for (int i = lane; i < SEQN * 64; i += 32) ps[i] = (bf16_t)0.f;
  for (int p = lane; p < SEQN; p += 32)
    mk[p] = (inp[base + (size_t)p * stride] < 10) ? 1 : 0;

  // Load q,k (with RoPE) and v (transposed) head tiles into LDS.
  for (int idx = lane; idx < SEQN * 8; idx += 32) {
    int p = idx >> 3, c = idx & 7;
    size_t t = base + (size_t)p * stride;
    const bf16_t* qg = qb + t * DIM_ + h * DH_ + c * 8;
    const bf16_t* kg = kb + t * DIM_ + h * DH_ + c * 8;
    const bf16_t* vg = vb + t * DIM_ + h * DH_ + c * 8;
    float qv[8], kv[8];
#pragma unroll
    for (int e = 0; e < 8; e++) { qv[e] = (float)qg[e]; kv[e] = (float)kg[e]; }
#pragma unroll
    for (int pr = 0; pr < 4; pr++) {
      int cidx = c * 4 + pr;                      // rotary pair 0..31
      float inv = __expf(-(float)cidx * 0.2878231366242557f); // 1e4^(-c/32)
      float ang = (float)p * inv;
      float sn, cs;
      __sincosf(ang, &sn, &cs);
      float a0 = qv[2 * pr], a1 = qv[2 * pr + 1];
      qv[2 * pr] = a0 * cs - a1 * sn;
      qv[2 * pr + 1] = a1 * cs + a0 * sn;
      float b0 = kv[2 * pr], b1 = kv[2 * pr + 1];
      kv[2 * pr] = b0 * cs - b1 * sn;
      kv[2 * pr + 1] = b1 * cs + b0 * sn;
    }
#pragma unroll
    for (int e = 0; e < 8; e++) {
      qs[p * 64 + c * 8 + e] = (bf16_t)qv[e];
      ks[p * 64 + c * 8 + e] = (bf16_t)kv[e];
      vt[(c * 8 + e) * 64 + p] = vg[e];
    }
  }
  __syncthreads();

  int half = lane >> 4, r = lane & 15;

  // scores = q . k^T, masked, * sqrt(DH)=8
  for (int mt = 0; mt < 3; mt++) {
    for (int nt = 0; nt < 3; nt++) {
      v8f acc;
#pragma unroll
      for (int e = 0; e < 8; e++) acc[e] = 0.f;
#pragma unroll
      for (int kk = 0; kk < 2; kk++) {
        bf16x16 a = frag_a(&qs[(mt * 16) * 64 + kk * 32], 64, half, r);
        bf16x16 b = frag_b(&ks[(nt * 16) * 64 + kk * 32], 64, half, r);
        acc = wmma_bf16(a, b, acc);
      }
      int n = nt * 16 + r;
      int mn = mk[n];
#pragma unroll
      for (int vi = 0; vi < 8; vi++) {
        int m = mt * 16 + vi + half * 8;
        float v = acc[vi] * 8.0f;
        sc[m * SEQN + n] = (mk[m] == mn) ? v : -3.0e38f;
      }
    }
  }
  __syncthreads();

  // row softmax -> bf16 probs
  for (int m = lane; m < SEQN; m += 32) {
    float mx = -3.4e38f;
    for (int n = 0; n < SEQN; n++) mx = fmaxf(mx, sc[m * SEQN + n]);
    float sum = 0.f;
    for (int n = 0; n < SEQN; n++) sum += __expf(sc[m * SEQN + n] - mx);
    float inv = 1.f / sum;
    for (int n = 0; n < SEQN; n++)
      ps[m * 64 + n] = (bf16_t)(__expf(sc[m * SEQN + n] - mx) * inv);
  }
  __syncthreads();

  // o = P . V
  for (int mt = 0; mt < 3; mt++) {
    for (int nt = 0; nt < 4; nt++) {
      v8f acc;
#pragma unroll
      for (int e = 0; e < 8; e++) acc[e] = 0.f;
#pragma unroll
      for (int kk = 0; kk < 2; kk++) {
        bf16x16 a = frag_a(&ps[(mt * 16) * 64 + kk * 32], 64, half, r);
        bf16x16 b = frag_b(&vt[(nt * 16) * 64 + kk * 32], 64, half, r);
        acc = wmma_bf16(a, b, acc);
      }
      int n = nt * 16 + r;
#pragma unroll
      for (int vi = 0; vi < 8; vi++) {
        int m = mt * 16 + vi + half * 8;
        size_t t = base + (size_t)m * stride;
        ob[t * DIM_ + h * DH_ + n] = (bf16_t)acc[vi];
      }
    }
  }
}

// ---------------------------------------------------------------------------
// Residual: x = l2n(x + alpha*ALPHAFS*(l2n(branch*brscale) - x))
// ---------------------------------------------------------------------------
__global__ __launch_bounds__(256)
void residual_kernel(float* __restrict__ x, bf16_t* __restrict__ xb,
                     const float* __restrict__ branch,
                     const float* __restrict__ alpha, float brscale) {
  __shared__ float red[256];
  size_t t = blockIdx.x;
  int tid = threadIdx.x;
  float bv[3], xv[3];
  float ss = 0.f;
#pragma unroll
  for (int k = 0; k < 3; k++) {
    int d = tid + k * 256;
    float b = branch[t * DIM_ + d] * brscale;
    bv[k] = b;
    xv[k] = x[t * DIM_ + d];
    ss += b * b;
  }
  ss = block_reduce_256(ss, red);
  float binv = 1.f / fmaxf(sqrtf(ss), EPS_);
  float yv[3];
  float ss2 = 0.f;
#pragma unroll
  for (int k = 0; k < 3; k++) {
    int d = tid + k * 256;
    float nb = bv[k] * binv;
    float y = xv[k] + alpha[d] * ALPHAFS * (nb - xv[k]);
    yv[k] = y;
    ss2 += y * y;
  }
  ss2 = block_reduce_256(ss2, red);
  float inv = 1.f / fmaxf(sqrtf(ss2), EPS_);
#pragma unroll
  for (int k = 0; k < 3; k++) {
    int d = tid + k * 256;
    float y = yv[k] * inv;
    x[t * DIM_ + d] = y;
    xb[t * DIM_ + d] = (bf16_t)y;
  }
}

// ---------------------------------------------------------------------------
// Logits: out[t,c] = (x[t,:] . wlog_n[c,:]) * s_logit[c]*sqrt(D), c<10
// ---------------------------------------------------------------------------
__global__ __launch_bounds__(256)
void logits_kernel(const float* __restrict__ x, const bf16_t* __restrict__ wl,
                   const float* __restrict__ s_logit, float* __restrict__ out) {
  __shared__ float red[256];
  size_t t = blockIdx.x;
  int tid = threadIdx.x;
  float xr[3];
#pragma unroll
  for (int k = 0; k < 3; k++) xr[k] = x[t * DIM_ + tid + k * 256];
  for (int c = 0; c < 10; c++) {
    float p = 0.f;
#pragma unroll
    for (int k = 0; k < 3; k++)
      p += xr[k] * (float)wl[(size_t)c * DIM_ + tid + k * 256];
    p = block_reduce_256(p, red);
    if (tid == 0) out[t * 10 + c] = p * s_logit[c] * SQRTD;
  }
}

// ---------------------------------------------------------------------------
// Host launcher
// ---------------------------------------------------------------------------
static void launch_gemm(int epi, const bf16_t* A, const bf16_t* Bw,
                        int M, int N, int K, bf16_t* outb, float* outf,
                        const float* cscale, const bf16_t* gate,
                        hipStream_t st) {
  dim3 g(N / 128, M / 128), b(256);
  switch (epi) {
    case 0: gemm_kernel<0><<<g, b, 0, st>>>(A, Bw, M, N, K, outb, outf, cscale, gate); break;
    case 1: gemm_kernel<1><<<g, b, 0, st>>>(A, Bw, M, N, K, outb, outf, cscale, gate); break;
    case 2: gemm_kernel<2><<<g, b, 0, st>>>(A, Bw, M, N, K, outb, outf, cscale, gate); break;
    case 3: gemm_kernel<3><<<g, b, 0, st>>>(A, Bw, M, N, K, outb, outf, cscale, gate); break;
    default: gemm_kernel<4><<<g, b, 0, st>>>(A, Bw, M, N, K, outb, outf, cscale, gate); break;
  }
}

extern "C" void kernel_launch(void* const* d_in, const int* in_sizes, int n_in,
                              void* d_out, int out_size, void* d_ws,
                              size_t ws_size, hipStream_t stream) {
  (void)in_sizes; (void)n_in; (void)out_size; (void)ws_size;

  const int* input_ = (const int*)d_in[0];
  const int* task = (const int*)d_in[1];
  const float* W_color = (const float*)d_in[2];
  const float* W_task = (const float*)d_in[3];
  const float* s_task = (const float*)d_in[4];
  const float* W_logit = (const float*)d_in[5];
  const float* s_logit = (const float*)d_in[6];

  struct Layer {
    const float *Wq, *Wk, *Wv, *Wo, *s_qk, *alpha_attn;
    const float *Wh, *Wg, *Wf, *s_hidden, *s_gate, *alpha_ff;
  } L[DEPTH_];
  for (int l = 0; l < DEPTH_; l++) {
    int b = 7 + l * 12;
    L[l].Wq = (const float*)d_in[b + 0];
    L[l].Wk = (const float*)d_in[b + 1];
    L[l].Wv = (const float*)d_in[b + 2];
    L[l].Wo = (const float*)d_in[b + 3];
    L[l].s_qk = (const float*)d_in[b + 4];
    L[l].alpha_attn = (const float*)d_in[b + 5];
    L[l].Wh = (const float*)d_in[b + 6];
    L[l].Wg = (const float*)d_in[b + 7];
    L[l].Wf = (const float*)d_in[b + 8];
    L[l].s_hidden = (const float*)d_in[b + 9];
    L[l].s_gate = (const float*)d_in[b + 10];
    L[l].alpha_ff = (const float*)d_in[b + 11];
  }

  // Workspace layout (~341 MB): normalized bf16 weights (resident in L2),
  // residual stream f32+bf16, q/k/v/o bf16, branch f32.  The SwiGLU gate /
  // hidden buffer (T x 2048 bf16) aliases the q/k/v region (attention for the
  // layer is complete before FF runs).
  char* base = (char*)d_ws;
  size_t off = 0;
  auto take = [&](size_t bytes) -> char* {
    char* r = base + off;
    off += (bytes + 255) & ~(size_t)255;
    return r;
  };

  bf16_t *wq_n[DEPTH_], *wk_n[DEPTH_], *wv_n[DEPTH_], *wo_n[DEPTH_];
  bf16_t *wh_n[DEPTH_], *wg_n[DEPTH_], *wf_n[DEPTH_];
  for (int l = 0; l < DEPTH_; l++) {
    wq_n[l] = (bf16_t*)take((size_t)DIM_ * DIM_ * 2);
    wk_n[l] = (bf16_t*)take((size_t)DIM_ * DIM_ * 2);
    wv_n[l] = (bf16_t*)take((size_t)DIM_ * DIM_ * 2);
    wo_n[l] = (bf16_t*)take((size_t)DIM_ * DIM_ * 2);
    wh_n[l] = (bf16_t*)take((size_t)FFI_ * DIM_ * 2);
    wg_n[l] = (bf16_t*)take((size_t)FFI_ * DIM_ * 2);
    wf_n[l] = (bf16_t*)take((size_t)DIM_ * FFI_ * 2);
  }
  float* wcol_n = (float*)take((size_t)11 * DIM_ * 4);
  float* wtask_n = (float*)take((size_t)400 * DIM_ * 4);
  bf16_t* wlog_n = (bf16_t*)take((size_t)10 * DIM_ * 2);
  float* x = (float*)take((size_t)T_TOK * DIM_ * 4);
  bf16_t* xb = (bf16_t*)take((size_t)T_TOK * DIM_ * 2);
  bf16_t* qb = (bf16_t*)take((size_t)T_TOK * DIM_ * 2);
  bf16_t* kb = (bf16_t*)take((size_t)T_TOK * DIM_ * 2);
  bf16_t* vb = (bf16_t*)take((size_t)T_TOK * DIM_ * 2);
  bf16_t* ob = (bf16_t*)take((size_t)T_TOK * DIM_ * 2);
  float* branch = (float*)take((size_t)T_TOK * DIM_ * 4);
  bf16_t* gbuf = qb;  // T x 2048 bf16, aliases q/k/v region (75.5MB < 84.9MB)

  // --- weight parametrization: L2-normalize rows, downconvert to bf16 ---
  for (int l = 0; l < DEPTH_; l++) {
    rownorm_bf16_kernel<<<DIM_, 256, 0, stream>>>(L[l].Wq, wq_n[l], DIM_);
    rownorm_bf16_kernel<<<DIM_, 256, 0, stream>>>(L[l].Wk, wk_n[l], DIM_);
    rownorm_bf16_kernel<<<DIM_, 256, 0, stream>>>(L[l].Wv, wv_n[l], DIM_);
    rownorm_bf16_kernel<<<DIM_, 256, 0, stream>>>(L[l].Wo, wo_n[l], DIM_);
    rownorm_bf16_kernel<<<FFI_, 256, 0, stream>>>(L[l].Wh, wh_n[l], DIM_);
    rownorm_bf16_kernel<<<FFI_, 256, 0, stream>>>(L[l].Wg, wg_n[l], DIM_);
    rownorm_bf16_kernel<<<DIM_, 256, 0, stream>>>(L[l].Wf, wf_n[l], FFI_);
  }
  rownorm_f32_kernel<<<11, 256, 0, stream>>>(W_color, wcol_n, DIM_);
  rownorm_f32_kernel<<<400, 256, 0, stream>>>(W_task, wtask_n, DIM_);
  rownorm_bf16_kernel<<<10, 256, 0, stream>>>(W_logit, wlog_n, DIM_);

  // --- embedding ---
  embed_kernel<<<T_TOK, 256, 0, stream>>>(input_, task, wcol_n, wtask_n,
                                          s_task, x, xb);

  // --- layers ---
  for (int l = 0; l < DEPTH_; l++) {
    // QKV projections
    launch_gemm(0, xb, wq_n[l], T_TOK, DIM_, DIM_, qb, nullptr, nullptr, nullptr, stream);
    launch_gemm(0, xb, wk_n[l], T_TOK, DIM_, DIM_, kb, nullptr, nullptr, nullptr, stream);
    launch_gemm(0, xb, wv_n[l], T_TOK, DIM_, DIM_, vb, nullptr, nullptr, nullptr, stream);
    // per-head l2norm + scale (shared by both axial directions)
    qkprep_kernel<<<(size_t)T_TOK * HEADS_, 32, 0, stream>>>(qb, kb, L[l].s_qk);
    // row-direction attention + Wo
    attn_kernel<<<384 * HEADS_, 32, 0, stream>>>(qb, kb, vb, input_, ob, 0);
    launch_gemm(3, ob, wo_n[l], T_TOK, DIM_, DIM_, nullptr, branch, nullptr, nullptr, stream);
    // col-direction attention + Wo (accumulate)
    attn_kernel<<<384 * HEADS_, 32, 0, stream>>>(qb, kb, vb, input_, ob, 1);
    launch_gemm(4, ob, wo_n[l], T_TOK, DIM_, DIM_, nullptr, branch, nullptr, nullptr, stream);
    // residual with (xi+xj)/2
    residual_kernel<<<T_TOK, 256, 0, stream>>>(x, xb, branch, L[l].alpha_attn, 0.5f);
    // SwiGLU FF: g = silu(x Wg^T * s_gate * sqrt(D)); hg = (x Wh^T * s_hidden) * g
    launch_gemm(1, xb, wg_n[l], T_TOK, FFI_, DIM_, gbuf, nullptr, L[l].s_gate, nullptr, stream);
    launch_gemm(2, xb, wh_n[l], T_TOK, FFI_, DIM_, gbuf, nullptr, L[l].s_hidden, gbuf, stream);
    launch_gemm(3, gbuf, wf_n[l], T_TOK, DIM_, FFI_, nullptr, branch, nullptr, nullptr, stream);
    residual_kernel<<<T_TOK, 256, 0, stream>>>(x, xb, branch, L[l].alpha_ff, 1.0f);
  }

  // --- logits ---
  logits_kernel<<<T_TOK, 256, 0, stream>>>(x, wlog_n, s_logit, (float*)d_out);
}
